// Model_40450001994317
// MI455X (gfx1250) — compile-verified, matrix operand order checked by reading, not run on hardware
//
#include <hip/hip_runtime.h>
#include <hip/hip_bf16.h>

typedef __attribute__((ext_vector_type(16))) __bf16 v16bf;
typedef __attribute__((ext_vector_type(8)))  __bf16 bf16x8;
typedef __attribute__((ext_vector_type(2)))  __bf16 bf16x2;
typedef __attribute__((ext_vector_type(8)))  float  v8f;

#define Bsz   1024
#define Hdim  256
#define Kdim  512            // 2*H
#define NCOLS 3328           // 6*256 (W1) + 6*256 (W2) + 256 (Wmerge)
#define BM    128
#define BN    64
#define BK    32

// ---------------------------------------------------------------------------
// Build concatenated bf16 weight matrix Wcat[k][n], k in [0,512), n in [0,3328)
//   n in [0,1536):    W1[i=n>>8][k][c=n&255]
//   n in [1536,3072): W2[i][k][c]
//   n in [3072,3328): Wmerge[k][c]
// ---------------------------------------------------------------------------
__global__ __launch_bounds__(256) void build_wcat(
    const float* __restrict__ W1, const float* __restrict__ W2,
    const float* __restrict__ Wm, __bf16* __restrict__ Wcat)
{
  int idx = blockIdx.x * 256 + threadIdx.x;
  if (idx >= Kdim * NCOLS) return;
  int k = idx / NCOLS;
  int n = idx - k * NCOLS;
  float v;
  if (n < 1536) {
    int i = n >> 8; int c = n & 255;
    v = W1[((size_t)i * Kdim + k) * Hdim + c];
  } else if (n < 3072) {
    int i = (n - 1536) >> 8; int c = n & 255;
    v = W2[((size_t)i * Kdim + k) * Hdim + c];
  } else {
    v = Wm[(size_t)k * Hdim + (n - 3072)];
  }
  Wcat[(size_t)k * NCOLS + n] = (__bf16)v;
}

// ---------------------------------------------------------------------------
// Fragment load from LDS row-major [row][32] tile, per CDNA5 16-bit A layout:
//   lane half h = lane>>4, m = lane&15
//   VGPR v in 0..3 : K = 8h + 2v, 8h + 2v + 1
//   VGPR v in 4..7 : K = 16 + 8h + 2(v-4), +1
// B is loaded identically from a transposed [n][32] tile (N in lane).
// ---------------------------------------------------------------------------
__device__ inline v16bf load_frag(const __bf16* lds, int rowBase, int ml, int h)
{
  const __bf16* p = lds + (size_t)(rowBase + ml) * BK;
  v16bf f;
#pragma unroll
  for (int v = 0; v < 8; ++v) {
    int k = (v < 4) ? (h * 8 + 2 * v) : (16 + h * 8 + 2 * (v - 4));
    bf16x2 t = *(const bf16x2*)(p + k);
    f[2 * v]     = t.x;
    f[2 * v + 1] = t.y;
  }
  return f;
}

// ---------------------------------------------------------------------------
// Per-cell GEMM:  Z[B,3328] = [hx | hy] (B x 512, bf16) @ Wcat (512 x 3328, bf16)
// f32 accumulation via v_wmma_f32_16x16x32_bf16.
// Grid: (NCOLS/BN, B/BM) = (52, 8); 256 threads = 8 waves; wave -> 32x32 tile.
// ---------------------------------------------------------------------------
__global__ __launch_bounds__(256) void gemm_cell(
    const __bf16* __restrict__ hx, const __bf16* __restrict__ hy,
    const __bf16* __restrict__ Wcat, float* __restrict__ Z)
{
  __shared__ __bf16 Alds[BM * BK];   // [row][k]
  __shared__ __bf16 Blds[BN * BK];   // [n][k]   (transposed tile)

  const int tid  = threadIdx.x;
  const int wave = tid >> 5;
  const int lane = tid & 31;
  const int bm   = blockIdx.y;                // 0..7
  const int bn   = blockIdx.x;                // 0..51
  const int waveM = (wave & 3) * 32;          // 0,32,64,96
  const int waveN = (wave >> 2) * 32;         // 0,32
  const int h  = lane >> 4;
  const int ml = lane & 15;

  v8f acc00 = {0.f,0.f,0.f,0.f,0.f,0.f,0.f,0.f};
  v8f acc01 = acc00, acc10 = acc00, acc11 = acc00;

  for (int k0 = 0; k0 < Kdim; k0 += BK) {
    // ---- stage A tile (128x32) : src is hx for k<256, hy for k>=256 ----
    {
      const __bf16* src = (k0 < Hdim) ? hx : hy;
      const int kb = k0 & (Hdim - 1);
      const int r = tid >> 1;
      const int c = (tid & 1) * 16;
      const int4* g = reinterpret_cast<const int4*>(src + (size_t)(bm * BM + r) * Hdim + kb + c);
      int4* s4 = reinterpret_cast<int4*>(&Alds[(size_t)r * BK + c]);
      s4[0] = g[0];
      s4[1] = g[1];
    }
    // ---- stage B tile (32x64) transposed into Blds[n][k] ----
    {
      const int k  = tid >> 3;           // 0..31
      const int n0 = (tid & 7) * 8;      // 0..56
      bf16x8 t = *reinterpret_cast<const bf16x8*>(
          Wcat + (size_t)(k0 + k) * NCOLS + (size_t)bn * BN + n0);
#pragma unroll
      for (int j = 0; j < 8; ++j)
        Blds[(size_t)(n0 + j) * BK + k] = t[j];
    }
    __syncthreads();

    v16bf a0 = load_frag(Alds, waveM,      ml, h);
    v16bf a1 = load_frag(Alds, waveM + 16, ml, h);
    v16bf b0 = load_frag(Blds, waveN,      ml, h);
    v16bf b1 = load_frag(Blds, waveN + 16, ml, h);

    acc00 = __builtin_amdgcn_wmma_f32_16x16x32_bf16(false, a0, false, b0, (short)0, acc00, false, false);
    acc01 = __builtin_amdgcn_wmma_f32_16x16x32_bf16(false, a0, false, b1, (short)0, acc01, false, false);
    acc10 = __builtin_amdgcn_wmma_f32_16x16x32_bf16(false, a1, false, b0, (short)0, acc10, false, false);
    acc11 = __builtin_amdgcn_wmma_f32_16x16x32_bf16(false, a1, false, b1, (short)0, acc11, false, false);

    __syncthreads();
  }

  // ---- store C per CDNA5 f32 C/D layout: VGPR r -> row base + 8h + r; lane%16 -> col
  const int rB = bm * BM + waveM + h * 8;
  const int cB = bn * BN + waveN + ml;
#pragma unroll
  for (int r = 0; r < 8; ++r) {
    Z[(size_t)(rB +  0 + r) * NCOLS + cB +  0] = acc00[r];
    Z[(size_t)(rB +  0 + r) * NCOLS + cB + 16] = acc01[r];
    Z[(size_t)(rB + 16 + r) * NCOLS + cB +  0] = acc10[r];
    Z[(size_t)(rB + 16 + r) * NCOLS + cB + 16] = acc11[r];
  }
}

// ---------------------------------------------------------------------------
// Pointwise GRU cell update: gathers the one-hot-selected slices of Z,
// applies tanh/sigmoid/merge, writes bf16 hidden (for next GEMMs) + f32 copy.
// ---------------------------------------------------------------------------
__global__ __launch_bounds__(256) void cell_update(
    const float* __restrict__ Z, const float* __restrict__ b1,
    const float* __restrict__ b2, const int* __restrict__ samples,
    int nx, int ny, int nxn, int hasx, int hasy,
    __bf16* __restrict__ hout, float* __restrict__ hcur)
{
  int idx = blockIdx.x * 256 + threadIdx.x;   // B*H threads
  int l = idx >> 8;
  int k = idx & 255;
  const float* zrow = Z + (size_t)l * NCOLS;
  float pre1 = 0.f, pre2 = 0.f;
  if (hasx) {
    int sx = samples[l * 64 + nxn * 8 + ny];
    pre1 += zrow[sx * 256 + k];
    pre2 += zrow[1536 + sx * 256 + k];
  }
  if (hasy) {
    int sy = samples[l * 64 + nx * 8 + (ny - 1)];
    pre1 += zrow[(3 + sy) * 256 + k];
    pre2 += zrow[1536 + (3 + sy) * 256 + k];
  }
  float merge = zrow[3072 + k];
  float ht = tanhf(pre1 + b1[k]);
  float u  = 1.f / (1.f + expf(-(pre2 + b2[k])));
  float hv = u * ht + (1.f - u) * merge;
  hout[idx] = (__bf16)hv;
  hcur[idx] = hv;
}

// ---------------------------------------------------------------------------
// Logits + autoregressive masking update: one thread per batch element.
// probs = softmax(h@Wl1+bl1); phase = pi*softsign(h@Wl2+bl2); mask by counts.
// ---------------------------------------------------------------------------
__global__ __launch_bounds__(256) void logits_update(
    const float* __restrict__ hcur,
    const float* __restrict__ Wl1, const float* __restrict__ bl1,
    const float* __restrict__ Wl2, const float* __restrict__ bl2,
    const int* __restrict__ samples, int nx, int ny, int kcell,
    float* __restrict__ n_up, float* __restrict__ n_dn,
    float* __restrict__ log_a, float* __restrict__ log_p)
{
  int l = blockIdx.x * 256 + threadIdx.x;
  if (l >= Bsz) return;
  float l1[3] = {bl1[0], bl1[1], bl1[2]};
  float l2[3] = {bl2[0], bl2[1], bl2[2]};
  const float* hp = hcur + (size_t)l * Hdim;
  for (int k = 0; k < Hdim; ++k) {
    float hv = hp[k];
#pragma unroll
    for (int c = 0; c < 3; ++c) {
      l1[c] += hv * Wl1[k * 3 + c];
      l2[c] += hv * Wl2[k * 3 + c];
    }
  }
  // softmax over 3
  float m = fmaxf(l1[0], fmaxf(l1[1], l1[2]));
  float e0 = expf(l1[0] - m), e1 = expf(l1[1] - m), e2 = expf(l1[2] - m);
  float es = e0 + e1 + e2;
  float p[3] = {e0 / es, e1 / es, e2 / es};
  const float PI = 3.14159265358979323846f;
  float ph[3];
#pragma unroll
  for (int c = 0; c < 3; ++c) ph[c] = PI * (l2[c] / (1.f + fabsf(l2[c])));

  float nup = n_up[l], ndn = n_dn[l];
  // bl_up = 24, bl_dn = 24, bl_hole = 16
  float mk[3];
  mk[2] = (24.f - nup > 0.f) ? 1.f : 0.f;
  mk[1] = (24.f - ndn > 0.f) ? 1.f : 0.f;
  mk[0] = (16.f - ((float)kcell - nup - ndn) > 0.f) ? 1.f : 0.f;
  float a0 = p[0] * mk[0], a1 = p[1] * mk[1], a2 = p[2] * mk[2];
  float asum = fmaxf(a0 + a1 + a2, 1e-30f);
  int s = samples[l * 64 + nx * 8 + ny];
  float asel = ((s == 0) ? a0 : (s == 1) ? a1 : a2) / asum;
  log_a[l] += logf(fmaxf(asel, 1e-12f));
  log_p[l] += ph[s];
  n_up[l] = nup + ((s == 2) ? 1.f : 0.f);
  n_dn[l] = ndn + ((s == 1) ? 1.f : 0.f);
}

__global__ __launch_bounds__(256) void finalize_out(
    const float* __restrict__ log_a, const float* __restrict__ log_p,
    float* __restrict__ out)
{
  int i = blockIdx.x * 256 + threadIdx.x;
  if (i >= Bsz) return;
  out[i]       = 0.5f * log_a[i];
  out[Bsz + i] = log_p[i];
}

// ---------------------------------------------------------------------------
extern "C" void kernel_launch(void* const* d_in, const int* in_sizes, int n_in,
                              void* d_out, int out_size, void* d_ws, size_t ws_size,
                              hipStream_t stream)
{
  const int*   samples = (const int*)  d_in[0];
  const float* W1      = (const float*)d_in[1];
  const float* b1      = (const float*)d_in[2];
  const float* W2      = (const float*)d_in[3];
  const float* b2      = (const float*)d_in[4];
  const float* Wm      = (const float*)d_in[5];
  const float* Wl1     = (const float*)d_in[6];
  const float* bl1     = (const float*)d_in[7];
  const float* Wl2     = (const float*)d_in[8];
  const float* bl2     = (const float*)d_in[9];
  float* out = (float*)d_out;

  // ---- workspace carving (256B aligned) ----
  size_t off = 0;
  char* base = (char*)d_ws;
  auto alloc = [&](size_t bytes) -> void* {
    void* p = base + off;
    off += (bytes + 255) & ~(size_t)255;
    return p;
  };
  __bf16* Wcat = (__bf16*)alloc((size_t)Kdim * NCOLS * 2);   // 3.25 MB
  float*  Z    = (float*) alloc((size_t)Bsz * NCOLS * 4);    // 13.0 MB
  __bf16* rows = (__bf16*)alloc((size_t)16 * Bsz * Hdim * 2);// 8.0 MB (2 rows x 8 cells)
  __bf16* zbuf = (__bf16*)alloc((size_t)Bsz * Hdim * 2);     // 0.5 MB
  float*  hcur = (float*) alloc((size_t)Bsz * Hdim * 4);     // 1.0 MB
  float*  nup  = (float*) alloc(Bsz * 4);
  float*  ndn  = (float*) alloc(Bsz * 4);
  float*  lga  = (float*) alloc(Bsz * 4);
  float*  lgp  = (float*) alloc(Bsz * 4);
  (void)ws_size; (void)in_sizes; (void)n_in; (void)out_size;

  // ---- per-call init (graph-capturable async memsets) ----
  hipMemsetAsync(zbuf, 0, (size_t)Bsz * Hdim * 2, stream);
  hipMemsetAsync(nup, 0, Bsz * 4, stream);
  hipMemsetAsync(ndn, 0, Bsz * 4, stream);
  hipMemsetAsync(lga, 0, Bsz * 4, stream);
  hipMemsetAsync(lgp, 0, Bsz * 4, stream);

  // ---- build bf16 concatenated weights ----
  {
    int total = Kdim * NCOLS;
    build_wcat<<<(total + 255) / 256, 256, 0, stream>>>(W1, W2, Wm, Wcat);
  }

  // ---- snake-order recurrence over the 8x8 lattice ----
  int kcell = 0;
  for (int ny = 0; ny < 8; ++ny) {
    const int dx = (ny % 2 == 0) ? -1 : 1;
    for (int t = 0; t < 8; ++t) {
      const int nx  = (ny % 2 == 0) ? t : (7 - t);
      const int nxn = nx + dx;
      const int hasx = (nxn >= 0 && nxn < 8) ? 1 : 0;
      const int hasy = (ny > 0) ? 1 : 0;
      __bf16* hx = hasx ? rows + ((size_t)((ny & 1) * 8 + nxn) * Bsz * Hdim) : zbuf;
      __bf16* hy = hasy ? rows + ((size_t)(((ny - 1) & 1) * 8 + nx) * Bsz * Hdim) : zbuf;
      __bf16* ho = rows + ((size_t)((ny & 1) * 8 + nx) * Bsz * Hdim);

      gemm_cell<<<dim3(NCOLS / BN, Bsz / BM), 256, 0, stream>>>(hx, hy, Wcat, Z);
      cell_update<<<(Bsz * Hdim) / 256, 256, 0, stream>>>(
          Z, b1, b2, samples, nx, ny, nxn, hasx, hasy, ho, hcur);
      logits_update<<<Bsz / 256, 256, 0, stream>>>(
          hcur, Wl1, bl1, Wl2, bl2, samples, nx, ny, kcell, nup, ndn, lga, lgp);
      ++kcell;
    }
  }

  finalize_out<<<Bsz / 256, 256, 0, stream>>>(lga, lgp, out);
}
